// MoELayer_53231824666804
// MI455X (gfx1250) — compile-verified, matrix operand order checked by reading, not run on hardware
//
#include <hip/hip_runtime.h>

// ---------------- problem constants ----------------
#define TN 2048           // tokens
#define TD 1024           // hidden dim
#define TI 2816           // intermediate dim
#define TE 8              // experts
#define SLOTS (TN * 2)    // top-2 -> 4096 routed slots, slot = 2*token + k

typedef __attribute__((ext_vector_type(16))) __bf16 v16bf;
typedef __attribute__((ext_vector_type(8)))  float  v8f;

__device__ __forceinline__ unsigned short f2bf(float f) {
    unsigned int u = __builtin_bit_cast(unsigned int, f);
    unsigned int r = u + 0x7FFFu + ((u >> 16) & 1u);   // round-to-nearest-even
    return (unsigned short)(r >> 16);
}

// ---------------- 0: zero dispatch counters ----------------
__global__ void moe_zero_kernel(int* cnt) {
    if (threadIdx.x < TE) cnt[threadIdx.x] = 0;
}

// ---------------- 1: x f32 -> bf16 ----------------
__global__ __launch_bounds__(256) void moe_cvt_x_kernel(const float* __restrict__ x,
                                                        unsigned short* __restrict__ xb) {
    int i = (blockIdx.x * 256 + threadIdx.x) * 4;
    float4 v = *(const float4*)(x + i);
    uint2 o;
    o.x = (unsigned)f2bf(v.x) | ((unsigned)f2bf(v.y) << 16);
    o.y = (unsigned)f2bf(v.z) | ((unsigned)f2bf(v.w) << 16);
    *(uint2*)(xb + i) = o;
}

// ---------------- 2: weight transpose + f32->bf16 ----------------
// src R x C (row-major f32) -> dst C x R (row-major bf16).  dst[c][r] = src[r][c]
// grid.z: 0..7 gate (R=D,C=I), 8..15 up (R=D,C=I), 16..23 down (R=I,C=D)
__global__ __launch_bounds__(256) void moe_wcvt_kernel(const float* __restrict__ wg,
                                                       const float* __restrict__ wu,
                                                       const float* __restrict__ wd,
                                                       unsigned short* __restrict__ wgb,
                                                       unsigned short* __restrict__ wub,
                                                       unsigned short* __restrict__ wdb) {
    int id = blockIdx.z;
    const float* src;
    unsigned short* dst;
    int R, C;
    if (id < 16) {
        R = TD; C = TI;
        int e = id & 7;
        src = (id < 8 ? wg : wu) + (size_t)e * TD * TI;
        dst = (id < 8 ? wgb : wub) + (size_t)e * TD * TI;
    } else {
        R = TI; C = TD;
        int e = id - 16;
        src = wd + (size_t)e * TI * TD;
        dst = wdb + (size_t)e * TI * TD;
    }
    int r0 = blockIdx.y * 32, c0 = blockIdx.x * 32;
    if (r0 >= R || c0 >= C) return;

    __shared__ float t[32][33];
    int tx = threadIdx.x & 31, ty = threadIdx.x >> 5;   // 32 x 8
#pragma unroll
    for (int j = 0; j < 4; ++j)
        t[ty + 8 * j][tx] = src[(size_t)(r0 + ty + 8 * j) * C + c0 + tx];
    __syncthreads();
#pragma unroll
    for (int j = 0; j < 4; ++j)
        dst[(size_t)(c0 + ty + 8 * j) * R + r0 + tx] = f2bf(t[tx][ty + 8 * j]);
}

// ---------------- 3: router + dispatch ----------------
__global__ __launch_bounds__(256) void moe_router_kernel(const float* __restrict__ x,
                                                         const float* __restrict__ wr,
                                                         float* __restrict__ probs,
                                                         float* __restrict__ slot_wt,
                                                         int* __restrict__ list_tok,
                                                         int* __restrict__ list_slot,
                                                         int* __restrict__ cnt) {
    __shared__ float wrs[TE * TD];                      // 32 KB
    int tid = threadIdx.x;
    for (int i = tid; i < TE * TD; i += 256) wrs[i] = wr[i];
    __syncthreads();

    int wave = tid >> 5, lane = tid & 31;
    int n = blockIdx.x * 8 + wave;

    float acc[TE];
#pragma unroll
    for (int e = 0; e < TE; ++e) acc[e] = 0.f;
    const float* xr = x + (size_t)n * TD;
    for (int it = 0; it < TD / 32; ++it) {
        float xv = xr[lane + 32 * it];
#pragma unroll
        for (int e = 0; e < TE; ++e) acc[e] += xv * wrs[e * TD + lane + 32 * it];
    }
#pragma unroll
    for (int off = 16; off; off >>= 1)
#pragma unroll
        for (int e = 0; e < TE; ++e) acc[e] += __shfl_xor(acc[e], off);

    if (lane == 0) {
        float mx = acc[0];
#pragma unroll
        for (int e = 1; e < TE; ++e) mx = fmaxf(mx, acc[e]);
        float p[TE], s = 0.f;
#pragma unroll
        for (int e = 0; e < TE; ++e) { p[e] = __expf(acc[e] - mx); s += p[e]; }
        float inv = 1.f / s;
#pragma unroll
        for (int e = 0; e < TE; ++e) { p[e] *= inv; probs[(size_t)n * TE + e] = p[e]; }

        int i0 = 0;
#pragma unroll
        for (int e = 1; e < TE; ++e) if (p[e] > p[i0]) i0 = e;   // ties: lowest idx
        int i1 = (i0 == 0) ? 1 : 0;
#pragma unroll
        for (int e = 0; e < TE; ++e) if (e != i0 && p[e] > p[i1]) i1 = e;

        float ws = p[i0] + p[i1];
        float w0 = p[i0] / ws, w1 = p[i1] / ws;
        slot_wt[2 * n]     = w0;
        slot_wt[2 * n + 1] = w1;
        int pos = atomicAdd(&cnt[i0], 1);
        list_tok[i0 * TN + pos]  = n;
        list_slot[i0 * TN + pos] = 2 * n;
        pos = atomicAdd(&cnt[i1], 1);
        list_tok[i1 * TN + pos]  = n;
        list_slot[i1 * TN + pos] = 2 * n + 1;
    }
}

// ---------------- 4: gate/up GEMM + SiLU -> h (bf16) ----------------
// M-tile = 32 rows (2 WMMA sub-tiles share every B tile -> 2x arith intensity)
// grid (64 M-tiles, 11 I-tiles of 256, 8 experts), 256 threads = 8 waves
__global__ __launch_bounds__(256) void moe_gemm1_kernel(const unsigned short* __restrict__ xb,
                                                        const unsigned short* __restrict__ wgb,
                                                        const unsigned short* __restrict__ wub,
                                                        const int* __restrict__ list_tok,
                                                        const int* __restrict__ list_slot,
                                                        const int* __restrict__ cnt,
                                                        unsigned short* __restrict__ hbuf) {
    constexpr int CH1 = 512;                            // K halfs per chunk (2 chunks)
    constexpr int AS1 = CH1 + 16;                       // 1056 B stride (32B mult)
    __shared__ __align__(32) unsigned short As[32 * AS1];   // 33.8 KB
    __shared__ int slotS[32];

    int e  = blockIdx.z;
    int cn = cnt[e];
    int m0 = blockIdx.x * 32;
    if (m0 >= cn) return;
    int valid = min(32, cn - m0);
    int tid = threadIdx.x;

    if (tid < 32) slotS[tid] = (tid < valid) ? list_slot[e * TN + m0 + tid] : 0;
    int srow = tid >> 3, sseg = tid & 7;                // 8 threads/row, 64 halfs each
    int stok = (srow < valid) ? list_tok[e * TN + m0 + srow] : -1;

    int wave = tid >> 5, lane = tid & 31;
    int nb = blockIdx.y * 256 + wave * 32;              // this wave: cols nb..nb+31 of I
    int mr = lane & 15, ko = (lane >> 4) * 16;
    const unsigned short* pgB = wgb + (((size_t)e * TI + nb + mr) * TD + ko);
    const unsigned short* puB = wub + (((size_t)e * TI + nb + mr) * TD + ko);

    v8f z8 = {0,0,0,0,0,0,0,0};
    v8f cg[2][2] = {{z8, z8}, {z8, z8}};                // [m-subtile][n-subtile]
    v8f cu[2][2] = {{z8, z8}, {z8, z8}};

    for (int c = 0; c < 2; ++c) {
        __syncthreads();                                // prev chunk fully consumed
        {
            uint4* dst = (uint4*)&As[srow * AS1 + sseg * 64];
            if (stok >= 0) {
                const uint4* src = (const uint4*)&xb[(size_t)stok * TD + c * CH1 + sseg * 64];
#pragma unroll
                for (int j = 0; j < 8; ++j) dst[j] = src[j];
            } else {
                uint4 zz = {0u, 0u, 0u, 0u};
#pragma unroll
                for (int j = 0; j < 8; ++j) dst[j] = zz;
            }
        }
        __syncthreads();

        const unsigned short* pa0 = &As[mr * AS1 + ko];
        const unsigned short* pa1 = &As[(mr + 16) * AS1 + ko];
        const unsigned short* pg0 = pgB + c * CH1;
        const unsigned short* pg1 = pg0 + (size_t)16 * TD;
        const unsigned short* pu0 = puB + c * CH1;
        const unsigned short* pu1 = pu0 + (size_t)16 * TD;

        for (int ks = 0; ks < CH1 / 32; ++ks) {         // 16 steps
            v16bf a0  = *(const v16bf*)(pa0 + ks * 32);
            v16bf a1  = *(const v16bf*)(pa1 + ks * 32);
            v16bf bg0 = *(const v16bf*)(pg0 + ks * 32);
            v16bf bg1 = *(const v16bf*)(pg1 + ks * 32);
            v16bf bu0 = *(const v16bf*)(pu0 + ks * 32);
            v16bf bu1 = *(const v16bf*)(pu1 + ks * 32);
            cg[0][0] = __builtin_amdgcn_wmma_f32_16x16x32_bf16(false, a0, false, bg0, (short)0, cg[0][0], false, false);
            cg[0][1] = __builtin_amdgcn_wmma_f32_16x16x32_bf16(false, a0, false, bg1, (short)0, cg[0][1], false, false);
            cu[0][0] = __builtin_amdgcn_wmma_f32_16x16x32_bf16(false, a0, false, bu0, (short)0, cu[0][0], false, false);
            cu[0][1] = __builtin_amdgcn_wmma_f32_16x16x32_bf16(false, a0, false, bu1, (short)0, cu[0][1], false, false);
            cg[1][0] = __builtin_amdgcn_wmma_f32_16x16x32_bf16(false, a1, false, bg0, (short)0, cg[1][0], false, false);
            cg[1][1] = __builtin_amdgcn_wmma_f32_16x16x32_bf16(false, a1, false, bg1, (short)0, cg[1][1], false, false);
            cu[1][0] = __builtin_amdgcn_wmma_f32_16x16x32_bf16(false, a1, false, bu0, (short)0, cu[1][0], false, false);
            cu[1][1] = __builtin_amdgcn_wmma_f32_16x16x32_bf16(false, a1, false, bu1, (short)0, cu[1][1], false, false);
        }
    }

    int lh = lane >> 4, nl = lane & 15;
#pragma unroll
    for (int ms = 0; ms < 2; ++ms) {
#pragma unroll
        for (int r = 0; r < 8; ++r) {
            int m = ms * 16 + r + 8 * lh;
            if (m < valid) {
                size_t slot = (size_t)slotS[m];
                float g0 = cg[ms][0][r], u0 = cu[ms][0][r];
                float g1 = cg[ms][1][r], u1 = cu[ms][1][r];
                float h0 = (g0 / (1.f + __expf(-g0))) * u0;   // silu(g)*u
                float h1 = (g1 / (1.f + __expf(-g1))) * u1;
                hbuf[slot * TI + nb + nl]      = f2bf(h0);
                hbuf[slot * TI + nb + 16 + nl] = f2bf(h1);
            }
        }
    }
}

// ---------------- 5: down GEMM -> y (f32, per slot) ----------------
// M-tile = 32 rows; grid (64 M-tiles, 4 D-tiles of 256, 8 experts)
__global__ __launch_bounds__(256) void moe_gemm2_kernel(const unsigned short* __restrict__ hbuf,
                                                        const unsigned short* __restrict__ wdb,
                                                        const int* __restrict__ list_slot,
                                                        const int* __restrict__ cnt,
                                                        float* __restrict__ ybuf) {
    constexpr int CH2 = 704;                            // K halfs per chunk (4 chunks)
    constexpr int AS2 = CH2 + 16;                       // 1440 B stride (32B mult)
    __shared__ __align__(32) unsigned short As[32 * AS2];   // 46.1 KB
    __shared__ int slotS[32];

    int e  = blockIdx.z;
    int cn = cnt[e];
    int m0 = blockIdx.x * 32;
    if (m0 >= cn) return;
    int valid = min(32, cn - m0);
    int tid = threadIdx.x;

    if (tid < 32) slotS[tid] = (tid < valid) ? list_slot[e * TN + m0 + tid] : -1;
    __syncthreads();

    int srow = tid >> 3, sseg = tid & 7;                // 8 threads/row, 88 halfs each
    int sslot = slotS[srow];

    int wave = tid >> 5, lane = tid & 31;
    int nb = blockIdx.y * 256 + wave * 32;              // cols of D
    int mr = lane & 15, ko = (lane >> 4) * 16;
    int lh = lane >> 4, nl = lane & 15;
    const unsigned short* pbB = wdb + (((size_t)e * TD + nb + mr) * TI + ko);

    v8f z8 = {0,0,0,0,0,0,0,0};
    v8f cd[2][2] = {{z8, z8}, {z8, z8}};

    for (int c = 0; c < 4; ++c) {
        {
            uint4* dst = (uint4*)&As[srow * AS2 + sseg * 88];
            if (sslot >= 0) {
                const uint4* src = (const uint4*)&hbuf[(size_t)sslot * TI + c * CH2 + sseg * 88];
#pragma unroll
                for (int j = 0; j < 11; ++j) dst[j] = src[j];
            } else {
                uint4 zz = {0u, 0u, 0u, 0u};
#pragma unroll
                for (int j = 0; j < 11; ++j) dst[j] = zz;
            }
        }
        __syncthreads();

        const unsigned short* pa0 = &As[mr * AS2 + ko];
        const unsigned short* pa1 = &As[(mr + 16) * AS2 + ko];
        const unsigned short* pb0 = pbB + c * CH2;
        const unsigned short* pb1 = pb0 + (size_t)16 * TI;
        for (int ks = 0; ks < CH2 / 32; ++ks) {         // 22 steps
            v16bf a0 = *(const v16bf*)(pa0 + ks * 32);
            v16bf a1 = *(const v16bf*)(pa1 + ks * 32);
            v16bf b0 = *(const v16bf*)(pb0 + ks * 32);
            v16bf b1 = *(const v16bf*)(pb1 + ks * 32);
            cd[0][0] = __builtin_amdgcn_wmma_f32_16x16x32_bf16(false, a0, false, b0, (short)0, cd[0][0], false, false);
            cd[0][1] = __builtin_amdgcn_wmma_f32_16x16x32_bf16(false, a0, false, b1, (short)0, cd[0][1], false, false);
            cd[1][0] = __builtin_amdgcn_wmma_f32_16x16x32_bf16(false, a1, false, b0, (short)0, cd[1][0], false, false);
            cd[1][1] = __builtin_amdgcn_wmma_f32_16x16x32_bf16(false, a1, false, b1, (short)0, cd[1][1], false, false);
        }
        __syncthreads();                                // before next chunk overwrite
    }

#pragma unroll
    for (int ms = 0; ms < 2; ++ms) {
#pragma unroll
        for (int r = 0; r < 8; ++r) {
            int m = ms * 16 + r + 8 * lh;
            if (m < valid) {
                size_t slot = (size_t)slotS[m];
                ybuf[slot * TD + nb + nl]      = cd[ms][0][r];
                ybuf[slot * TD + nb + 16 + nl] = cd[ms][1][r];
            }
        }
    }
}

// ---------------- 6: combine (deterministic, no atomics) ----------------
__global__ __launch_bounds__(256) void moe_combine_kernel(const float* __restrict__ ybuf,
                                                          const float* __restrict__ slot_wt,
                                                          float* __restrict__ out) {
    int n  = blockIdx.x;
    int d0 = threadIdx.x * 4;
    float w0 = slot_wt[2 * n], w1 = slot_wt[2 * n + 1];
    float4 a = *(const float4*)&ybuf[(size_t)(2 * n) * TD + d0];
    float4 b = *(const float4*)&ybuf[(size_t)(2 * n + 1) * TD + d0];
    float4 o;
    o.x = w0 * a.x + w1 * b.x;
    o.y = w0 * a.y + w1 * b.y;
    o.z = w0 * a.z + w1 * b.z;
    o.w = w0 * a.w + w1 * b.w;
    *(float4*)&out[(size_t)n * TD + d0] = o;
}

// ---------------- 7: aux loss (deterministic tree reduce) ----------------
__global__ __launch_bounds__(256) void moe_aux_kernel(const float* __restrict__ probs,
                                                      const int* __restrict__ cnt,
                                                      float* __restrict__ out_aux) {
    __shared__ float sm[256][TE];
    int tid = threadIdx.x;
    float ps[TE];
#pragma unroll
    for (int e = 0; e < TE; ++e) ps[e] = 0.f;
    for (int n = tid; n < TN; n += 256)
#pragma unroll
        for (int e = 0; e < TE; ++e) ps[e] += probs[(size_t)n * TE + e];
#pragma unroll
    for (int e = 0; e < TE; ++e) sm[tid][e] = ps[e];
    __syncthreads();
    for (int s = 128; s; s >>= 1) {
        if (tid < s)
#pragma unroll
            for (int e = 0; e < TE; ++e) sm[tid][e] += sm[tid + s][e];
        __syncthreads();
    }
    if (tid == 0) {
        float aux = 0.f;
#pragma unroll
        for (int e = 0; e < TE; ++e) {
            float f = (float)cnt[e] / (float)(TN * 2);
            float p = sm[0][e] / (float)TN;
            aux += f * p;
        }
        out_aux[0] = (float)TE * aux;
    }
}

// ---------------- launch ----------------
extern "C" void kernel_launch(void* const* d_in, const int* in_sizes, int n_in,
                              void* d_out, int out_size, void* d_ws, size_t ws_size,
                              hipStream_t stream) {
    const float* x  = (const float*)d_in[0];   // (1,2048,1024)
    const float* wr = (const float*)d_in[1];   // (8,1024)
    const float* wg = (const float*)d_in[2];   // (8,1024,2816)
    const float* wu = (const float*)d_in[3];   // (8,1024,2816)
    const float* wd = (const float*)d_in[4];   // (8,2816,1024)
    float* out = (float*)d_out;                // 2048*1024 + 1 (aux)

    char* ws = (char*)d_ws;
    size_t off = 0;
    auto alloc = [&](size_t bytes) { size_t o = off; off += (bytes + 255) & ~(size_t)255; return o; };

    unsigned short* wgb  = (unsigned short*)(ws + alloc((size_t)TE * TD * TI * 2));
    unsigned short* wub  = (unsigned short*)(ws + alloc((size_t)TE * TD * TI * 2));
    unsigned short* wdb  = (unsigned short*)(ws + alloc((size_t)TE * TI * TD * 2));
    unsigned short* xb   = (unsigned short*)(ws + alloc((size_t)TN * TD * 2));
    unsigned short* hbuf = (unsigned short*)(ws + alloc((size_t)SLOTS * TI * 2));
    float*          ybuf = (float*)(ws + alloc((size_t)SLOTS * TD * 4));
    float*          probs   = (float*)(ws + alloc((size_t)TN * TE * 4));
    float*          slot_wt = (float*)(ws + alloc((size_t)SLOTS * 4));
    int*            list_tok  = (int*)(ws + alloc((size_t)TE * TN * 4));
    int*            list_slot = (int*)(ws + alloc((size_t)TE * TN * 4));
    int*            cnt       = (int*)(ws + alloc((size_t)TE * 4));
    (void)ws_size; (void)in_sizes; (void)n_in; (void)out_size;

    moe_zero_kernel<<<1, 32, 0, stream>>>(cnt);
    moe_cvt_x_kernel<<<(TN * TD) / (256 * 4), 256, 0, stream>>>(x, xb);
    moe_wcvt_kernel<<<dim3(TI / 32, TI / 32, 24), 256, 0, stream>>>(wg, wu, wd, wgb, wub, wdb);
    moe_router_kernel<<<TN / 8, 256, 0, stream>>>(x, wr, probs, slot_wt, list_tok, list_slot, cnt);
    moe_gemm1_kernel<<<dim3(TN / 32, TI / 256, TE), 256, 0, stream>>>(xb, wgb, wub, list_tok, list_slot, cnt, hbuf);
    moe_gemm2_kernel<<<dim3(TN / 32, TD / 256, TE), 256, 0, stream>>>(hbuf, wdb, list_slot, cnt, ybuf);
    moe_combine_kernel<<<TN, 256, 0, stream>>>(ybuf, slot_wt, out);
    moe_aux_kernel<<<1, 256, 0, stream>>>(probs, cnt, out + (size_t)TN * TD);
}
